// LocalSelfAttention_4020089389256
// MI455X (gfx1250) — compile-verified, matrix operand order checked by reading, not run on hardware
//
#include <hip/hip_runtime.h>

// Mask is eye(N) -> softmax is exactly one-hot -> attention output == V.
// Whole op == (x @ Wv + bv) @ W_out + b_out : two fused row-wise GEMMs.
// bf16 WMMA (v_wmma_f32_16x16x32_bf16) with fp32 accumulation; LDS-tiled.
// Weights are pre-converted to bf16 and pre-swizzled into WMMA B-fragment
// lane order in d_ws by a tiny prep kernel, so the GEMM hot loop does one
// contiguous 32B load per lane per K-step instead of 16 strided b32 gathers.

typedef __attribute__((ext_vector_type(16))) __bf16 v16bf;
typedef __attribute__((ext_vector_type(8)))  __bf16 v8bf;
typedef __attribute__((ext_vector_type(4)))  __bf16 v4bf;
typedef __attribute__((ext_vector_type(8)))  float  v8f;

#define BATCH      32
#define SEQ        1024
#define CDIM       768
#define ROWS_TOTAL (BATCH * SEQ)   // 32768
#define MTILE      64              // rows per workgroup
#define LSTR       776             // LDS row stride in bf16 (768 + 8 pad -> conflict-free)
#define NT_COUNT   48              // 768 / 16 output column tiles
#define KSTEPS     24              // 768 / 32 K steps
#define WPK_ELEMS  ((size_t)NT_COUNT * KSTEPS * 32 * 16)   // 589824 bf16 per matrix

#define WMMA_BF16(A, B, Cacc) \
    __builtin_amdgcn_wmma_f32_16x16x32_bf16(false, (A), false, (B), (short)0, (Cacc), false, false)

// ---- Fallback B-fragment: strided fp32 gather + in-register cvt ----
__device__ __forceinline__ v16bf load_w_frag(const float* __restrict__ W, int ldw,
                                             int ncol, int K0, int hb) {
    const float* p  = W + (size_t)(K0 + 8 * hb) * ldw + ncol;
    const float* p2 = p + (size_t)16 * ldw;
    v16bf f;
#pragma unroll
    for (int e = 0; e < 8; ++e) {
        f[e]     = (__bf16)p[(size_t)e * ldw];
        f[e + 8] = (__bf16)p2[(size_t)e * ldw];
    }
    return f;
}

// A-fragment: lane lp holds matrix row; two aligned 16B LDS reads per lane.
__device__ __forceinline__ v16bf load_a_frag(const __bf16* row, int K0, int hb) {
    const v8bf lo = *(const v8bf*)(row + K0 + 8 * hb);
    const v8bf hi = *(const v8bf*)(row + K0 + 8 * hb + 16);
    v16bf a;
#pragma unroll
    for (int e = 0; e < 8; ++e) { a[e] = lo[e]; a[e + 8] = hi[e]; }
    return a;
}

// ---- Prep: fp32 weights -> bf16, pre-swizzled into B-fragment lane order.
// dst fragment index t = (nt*KSTEPS + kk)*32 + lane; each thread emits one
// lane's v16bf (32B contiguous store).
__global__ __launch_bounds__(256)
void pack_weights(const float* __restrict__ W, int ldw, int colOff,
                  __bf16* __restrict__ dst) {
    const int t = blockIdx.x * 256 + threadIdx.x;
    if (t >= NT_COUNT * KSTEPS * 32) return;
    const int lane = t & 31;
    const int kk   = (t >> 5) % KSTEPS;
    const int nt   = t / (32 * KSTEPS);
    const int lp   = lane & 15;
    const int hb   = lane >> 4;
    const int ncol = colOff + nt * 16 + lp;
    const int K0   = kk * 32;
    v16bf f;
#pragma unroll
    for (int e = 0; e < 8; ++e) {
        f[e]     = (__bf16)W[(size_t)(K0 + 8 * hb + e) * ldw + ncol];
        f[e + 8] = (__bf16)W[(size_t)(K0 + 8 * hb + 16 + e) * ldw + ncol];
    }
    *(v16bf*)(dst + (size_t)t * 16) = f;
}

template <bool PACKED>
__global__ __launch_bounds__(256)
void fused_local_attn(const float* __restrict__ x,
                      const float* __restrict__ Wqkv,
                      const float* __restrict__ bqkv,
                      const float* __restrict__ Wout,
                      const float* __restrict__ bout,
                      const __bf16* __restrict__ wv_pk,
                      const __bf16* __restrict__ wo_pk,
                      float* __restrict__ out) {
    extern __shared__ __bf16 smem[];
    __bf16* xs = smem;                       // [MTILE][LSTR] x tile (bf16)
    __bf16* vs = smem + MTILE * LSTR;        // [MTILE][LSTR] V tile (bf16)

    const int tid  = threadIdx.x;
    const int wave = tid >> 5;
    const int lane = tid & 31;
    const int lp   = lane & 15;              // row/col within 16-wide fragment
    const int hb   = lane >> 4;              // half selector (K sub-block)
    const long row_base = (long)blockIdx.x * MTILE;

    // ---- Stage x tile: fp32 global -> bf16 LDS (fully coalesced float4) ----
    {
        const float4* src = (const float4*)(x + row_base * CDIM);
        for (int j = 0; j < 48; ++j) {
            const int i   = tid + j * 256;   // float4 index in tile, 0..12287
            const int row = i / 192;         // 192 float4 per row
            const int c4  = i - row * 192;
            const float4 v = src[i];
            v4bf pk;
            pk[0] = (__bf16)v.x; pk[1] = (__bf16)v.y;
            pk[2] = (__bf16)v.z; pk[3] = (__bf16)v.w;
            *(v4bf*)(xs + row * LSTR + c4 * 4) = pk;
        }
    }
    __syncthreads();

    // ---- GEMM 1: V = x @ Wqkv[:, 2C:3C] + bqkv[2C:3C]  -> vs (bf16 LDS) ----
    {
        const float* Wv = Wqkv + 2 * CDIM;   // column window into [C][3C]
        const float* bv = bqkv + 2 * CDIM;
        for (int nt = wave; nt < NT_COUNT; nt += 8) {
            const int ncol = nt * 16 + lp;
            const float bias = bv[ncol];
            v8f acc0, acc1, acc2, acc3;
#pragma unroll
            for (int r = 0; r < 8; ++r) { acc0[r] = bias; acc1[r] = bias; acc2[r] = bias; acc3[r] = bias; }
            for (int kk = 0; kk < KSTEPS; ++kk) {
                const int K0 = kk * 32;
                v16bf bf;
                if (PACKED) {
                    const size_t fi = ((size_t)(nt * KSTEPS + kk) * 32 + lane) * 16;
                    bf = *(const v16bf*)(wv_pk + fi);
                    if (kk + 1 < KSTEPS)
                        __builtin_prefetch(wv_pk + fi + 32 * 16, 0, 1);
                } else {
                    bf = load_w_frag(Wv, 3 * CDIM, ncol, K0, hb);
                    if (kk + 1 < KSTEPS)
                        __builtin_prefetch(Wv + (size_t)(K0 + 32) * (3 * CDIM) + ncol, 0, 1);
                }
                v16bf a;
                a = load_a_frag(xs + (lp     ) * LSTR, K0, hb); acc0 = WMMA_BF16(a, bf, acc0);
                a = load_a_frag(xs + (lp + 16) * LSTR, K0, hb); acc1 = WMMA_BF16(a, bf, acc1);
                a = load_a_frag(xs + (lp + 32) * LSTR, K0, hb); acc2 = WMMA_BF16(a, bf, acc2);
                a = load_a_frag(xs + (lp + 48) * LSTR, K0, hb); acc3 = WMMA_BF16(a, bf, acc3);
            }
#pragma unroll
            for (int r = 0; r < 8; ++r) {
                const int mr = r + 8 * hb;   // C/D layout: M = vgpr + 8*half, N = lane%16
                vs[(mr     ) * LSTR + ncol] = (__bf16)acc0[r];
                vs[(mr + 16) * LSTR + ncol] = (__bf16)acc1[r];
                vs[(mr + 32) * LSTR + ncol] = (__bf16)acc2[r];
                vs[(mr + 48) * LSTR + ncol] = (__bf16)acc3[r];
            }
        }
    }
    __syncthreads();

    // ---- GEMM 2: out = V @ W_out + b_out  -> global fp32 ----
    {
        float* orow = out + row_base * CDIM;
        for (int nt = wave; nt < NT_COUNT; nt += 8) {
            const int ncol = nt * 16 + lp;
            const float bias = bout[ncol];
            v8f acc0, acc1, acc2, acc3;
#pragma unroll
            for (int r = 0; r < 8; ++r) { acc0[r] = bias; acc1[r] = bias; acc2[r] = bias; acc3[r] = bias; }
            for (int kk = 0; kk < KSTEPS; ++kk) {
                const int K0 = kk * 32;
                v16bf bf;
                if (PACKED) {
                    const size_t fi = ((size_t)(nt * KSTEPS + kk) * 32 + lane) * 16;
                    bf = *(const v16bf*)(wo_pk + fi);
                    if (kk + 1 < KSTEPS)
                        __builtin_prefetch(wo_pk + fi + 32 * 16, 0, 1);
                } else {
                    bf = load_w_frag(Wout, CDIM, ncol, K0, hb);
                    if (kk + 1 < KSTEPS)
                        __builtin_prefetch(Wout + (size_t)(K0 + 32) * CDIM + ncol, 0, 1);
                }
                v16bf a;
                a = load_a_frag(vs + (lp     ) * LSTR, K0, hb); acc0 = WMMA_BF16(a, bf, acc0);
                a = load_a_frag(vs + (lp + 16) * LSTR, K0, hb); acc1 = WMMA_BF16(a, bf, acc1);
                a = load_a_frag(vs + (lp + 32) * LSTR, K0, hb); acc2 = WMMA_BF16(a, bf, acc2);
                a = load_a_frag(vs + (lp + 48) * LSTR, K0, hb); acc3 = WMMA_BF16(a, bf, acc3);
            }
#pragma unroll
            for (int r = 0; r < 8; ++r) {
                const int mr = r + 8 * hb;
                orow[(long)(mr     ) * CDIM + ncol] = acc0[r];
                orow[(long)(mr + 16) * CDIM + ncol] = acc1[r];
                orow[(long)(mr + 32) * CDIM + ncol] = acc2[r];
                orow[(long)(mr + 48) * CDIM + ncol] = acc3[r];
            }
        }
    }
}

extern "C" void kernel_launch(void* const* d_in, const int* in_sizes, int n_in,
                              void* d_out, int out_size, void* d_ws, size_t ws_size,
                              hipStream_t stream) {
    (void)in_sizes; (void)n_in; (void)out_size;
    const float* x    = (const float*)d_in[0];
    const float* Wqkv = (const float*)d_in[1];
    const float* bqkv = (const float*)d_in[2];
    const float* Wout = (const float*)d_in[3];
    const float* bout = (const float*)d_in[4];
    float* out = (float*)d_out;

    const int    nblocks   = ROWS_TOTAL / MTILE;                 // 512 workgroups
    const size_t lds_bytes = (size_t)2 * MTILE * LSTR * 2;       // ~194 KB (< 320 KB/WGP)
    const size_t ws_need   = 2 * WPK_ELEMS * 2;                  // ~2.25 MB

    if (d_ws && ws_size >= ws_need) {
        __bf16* wv_pk = (__bf16*)d_ws;
        __bf16* wo_pk = wv_pk + WPK_ELEMS;
        const int pblocks = (NT_COUNT * KSTEPS * 32 + 255) / 256;  // 144
        pack_weights<<<pblocks, 256, 0, stream>>>(Wqkv, 3 * CDIM, 2 * CDIM, wv_pk);
        pack_weights<<<pblocks, 256, 0, stream>>>(Wout, CDIM, 0, wo_pk);
        fused_local_attn<true><<<nblocks, 256, lds_bytes, stream>>>(
            x, Wqkv, bqkv, Wout, bout, wv_pk, wo_pk, out);
    } else {
        fused_local_attn<false><<<nblocks, 256, lds_bytes, stream>>>(
            x, Wqkv, bqkv, Wout, bout, nullptr, nullptr, out);
    }
}
